// ObservedAttentionScores_15960098472446
// MI455X (gfx1250) — 8-bit matrix operand order repaired and checked by reading, compiled, not run on hardware
//
#include <hip/hip_runtime.h>

#define B_   4
#define D_   64
#define H_   16
#define N_   2048
#define BH_  (B_ * H_)
#define FP8_MAX 448.0f

typedef __attribute__((ext_vector_type(8))) int   v8i;
typedef __attribute__((ext_vector_type(8))) float v8f;

__device__ __forceinline__ unsigned f32_to_e4m3(float x) {
#if __has_builtin(__builtin_amdgcn_cvt_pk_fp8_f32)
    return (unsigned)__builtin_amdgcn_cvt_pk_fp8_f32(x, x, 0, false) & 0xFFu;
#else
    unsigned u = __float_as_uint(x);
    unsigned s = (u >> 31) << 7;
    unsigned b = u & 0x7FFFFFFFu;
    float ax = __uint_as_float(b);
    if (ax != ax) return s | 0x7Fu;
    if (ax >= FP8_MAX) return s | 0x7Eu;
    unsigned lsb = (b >> 20) & 1u;
    b += 0x7FFFFu + lsb;
    int e = (int)(b >> 23) - 127;
    unsigned m = (b >> 20) & 7u;
    if (e < -6) {
        int q = (int)rintf(ax * 512.0f);
        if (q >= 8) return s | 0x08u;
        return s | (unsigned)q;
    }
    return s | ((unsigned)(e + 7) << 3) | m;
#endif
}

__global__ void init_amax_kernel(unsigned* amax_bits) {
    if (threadIdx.x < 2) amax_bits[threadIdx.x] = 0u;
}

__global__ void __launch_bounds__(256)
amax_kernel(const float* __restrict__ x, unsigned* __restrict__ amax_bits, int n4) {
    const float4* p = (const float4*)x;
    float m = 0.0f;
    for (int i = blockIdx.x * blockDim.x + threadIdx.x; i < n4;
         i += gridDim.x * blockDim.x) {
        float4 v = p[i];
        m = fmaxf(m, fmaxf(fmaxf(fabsf(v.x), fabsf(v.y)),
                           fmaxf(fabsf(v.z), fabsf(v.w))));
    }
    #pragma unroll
    for (int off = 16; off > 0; off >>= 1)
        m = fmaxf(m, __shfl_xor(m, off, 32));
    if ((threadIdx.x & 31) == 0)
        atomicMax(amax_bits, __float_as_uint(m));
}

__global__ void __launch_bounds__(256)
quant_kernel(const float* __restrict__ src, unsigned char* __restrict__ dst,
             const float* __restrict__ amax) {
    int tid = blockIdx.x * blockDim.x + threadIdx.x;
    int n  = tid & (N_ - 1);
    int bh = tid >> 11;
    int b = bh >> 4, h = bh & 15;
    float scale = FP8_MAX / fmaxf(amax[0], 1e-12f);
    const float* p = src + ((size_t)b * D_ * H_ + h) * N_ + n;
    unsigned dw[16];
    #pragma unroll
    for (int dg = 0; dg < 16; ++dg) {
        unsigned w = 0;
        #pragma unroll
        for (int j = 0; j < 4; ++j) {
            float xv = p[(size_t)(dg * 4 + j) * (H_ * N_)] * scale;
            xv = fminf(fmaxf(xv, -FP8_MAX), FP8_MAX);
            w |= (f32_to_e4m3(xv) & 0xFFu) << (8 * j);
        }
        dw[dg] = w;
    }
    uint4* o = (uint4*)(dst + (size_t)tid * D_);
    o[0] = make_uint4(dw[0],  dw[1],  dw[2],  dw[3]);
    o[1] = make_uint4(dw[4],  dw[5],  dw[6],  dw[7]);
    o[2] = make_uint4(dw[8],  dw[9],  dw[10], dw[11]);
    o[3] = make_uint4(dw[12], dw[13], dw[14], dw[15]);
}

__global__ void __launch_bounds__(256)
gemm_fp8_kernel(const unsigned char* __restrict__ qf8,
                const unsigned char* __restrict__ kf8,
                const float* __restrict__ amax,
                float* __restrict__ out) {
    const int tileM = blockIdx.x;
    const int tileN = blockIdx.y;
    const int bh    = blockIdx.z;
    const int lane  = threadIdx.x & 31;
    const int wave  = threadIdx.x >> 5;
    const int half  = lane >> 4;
    const int lid   = lane & 15;

    const float aq = fmaxf(amax[0], 1e-12f);
    const float ak = fmaxf(amax[1], 1e-12f);
    const float osc = (aq * ak) * (1.0f / (FP8_MAX * FP8_MAX * 8.0f));

    const unsigned char* Q = qf8 + (size_t)bh * N_ * D_;
    const unsigned char* K = kf8 + (size_t)bh * N_ * D_;

    const int n0 = tileN * 128 + wave * 16;

    const int2* ap = (const int2*)(Q + (size_t)(n0 + lid) * D_);
    int2 a0 = ap[0 + 2 * half];
    int2 a1 = ap[1 + 2 * half];
    int2 a2 = ap[4 + 2 * half];
    int2 a3 = ap[5 + 2 * half];
    v8i A = (v8i){a0.x, a0.y, a1.x, a1.y, a2.x, a2.y, a3.x, a3.y};

    const size_t outRow = ((size_t)bh * N_ + n0) * N_;

    #pragma unroll
    for (int mt = 0; mt < 8; ++mt) {
        const int m0 = tileM * 128 + mt * 16;
        const unsigned char* bp = K + (size_t)(m0 + lid) * D_ + half * 16;
        int4 b0 = *(const int4*)(bp);
        int4 b1 = *(const int4*)(bp + 32);
        v8i Bf = (v8i){b0.x, b0.y, b0.z, b0.w, b1.x, b1.y, b1.z, b1.w};

        v8f C = {};
        C = __builtin_amdgcn_wmma_f32_16x16x64_fp8_fp8(A, Bf, (short)0, C,
                                                       false, false);
        #pragma unroll
        for (int r = 0; r < 8; ++r) {
            __builtin_nontemporal_store(
                C[r] * osc,
                out + outRow + (size_t)(r + 8 * half) * N_ + (m0 + lid));
        }
    }
}

extern "C" void kernel_launch(void* const* d_in, const int* in_sizes, int n_in,
                              void* d_out, int out_size, void* d_ws, size_t ws_size,
                              hipStream_t stream) {
    (void)in_sizes; (void)n_in; (void)out_size; (void)ws_size;
    const float* q = (const float*)d_in[0];
    const float* k = (const float*)d_in[1];
    float* out = (float*)d_out;

    unsigned* amax_bits = (unsigned*)d_ws;
    float*    amax      = (float*)d_ws;
    unsigned char* qf8  = (unsigned char*)d_ws + 1024;
    unsigned char* kf8  = qf8 + (size_t)BH_ * N_ * D_;

    init_amax_kernel<<<1, 32, 0, stream>>>(amax_bits);

    const int n4 = (B_ * D_ * H_ * N_) / 4;
    amax_kernel<<<1024, 256, 0, stream>>>(q, amax_bits + 0, n4);
    amax_kernel<<<1024, 256, 0, stream>>>(k, amax_bits + 1, n4);

    const int rows = BH_ * N_;
    quant_kernel<<<rows / 256, 256, 0, stream>>>(q, qf8, amax + 0);
    quant_kernel<<<rows / 256, 256, 0, stream>>>(k, kf8, amax + 1);

    dim3 grid(N_ / 128, N_ / 128, BH_);
    gemm_fp8_kernel<<<grid, 256, 0, stream>>>(qf8, kf8, amax, out);
}
